// tensor_reconstruction_46282567581934
// MI455X (gfx1250) — matrix operand order repaired and checked by reading, not compiled, not run on hardware
//
#include <hip/hip_runtime.h>

// ---------------------------------------------------------------------------
// out = sum_{ijk} (X[i,j,k] - sum_r A[i,r] B[j,r] C[k,r])^2
//     = sum X^2 - 2 * sum(X .* recon) + sum recon^2
//
//   sum(X.*recon) : fused GEMM  Y = X_flat(262144 x 512) @ C(512 x 64)
//                   weighted by A[i,r]*B[j,r] and reduced, X read exactly once
//   sum recon^2   : sum_{rs} (A^T A) .* (B^T B) .* (C^T C)   (64x64 Gramians)
//
// bf16x3 split (hi + residual-lo, keep hh+hl+lh) with f32 WMMA accumulators
// gives ~fp32 accuracy on v_wmma_f32_16x16x32_bf16, the fast matrix path.
// ---------------------------------------------------------------------------

typedef __attribute__((ext_vector_type(16))) __bf16 v16bf;
typedef __attribute__((ext_vector_type(8)))  float  v8f;

#define IJ_ROWS   (512 * 512)   // flattened (i,j)
#define K_DIM     512
#define R_DIM     64
#define ROWS_PER_WAVE  16
#define WAVES_PER_BLK  8
#define ROWS_PER_BLK   (ROWS_PER_WAVE * WAVES_PER_BLK)   // 128

// ws layout:
//   [0 .. 15]                 float acc[4]: 0=sumX2, 1=sumXrecon, 2=sumRecon2
//   [64 .. 64+65536)          CtHi : __bf16 [64][512]  (C transposed, hi part)
//   [64+65536 .. 64+131072)   CtLo : __bf16 [64][512]  (residual part)
#define WS_CT_OFF   64
#define CT_BYTES    (R_DIM * K_DIM * 2)

// ---- prep: Ct[r][k] = split(C[k][r]) into bf16 hi + bf16 residual ---------
__global__ void ct_split_kernel(const float* __restrict__ C,
                                __bf16* __restrict__ CtHi,
                                __bf16* __restrict__ CtLo) {
    int idx = blockIdx.x * blockDim.x + threadIdx.x;   // 0 .. 64*512-1
    int r = idx >> 9;
    int k = idx & 511;
    float c  = C[k * R_DIM + r];
    __bf16 h = (__bf16)c;
    CtHi[idx] = h;
    CtLo[idx] = (__bf16)(c - (float)h);
}

// ---- sum recon^2 = sum_{rs} (A^T A)(B^T B)(C^T C) -------------------------
__global__ __launch_bounds__(64)
void gram_kernel(const float* __restrict__ A, const float* __restrict__ B,
                 const float* __restrict__ C, float* __restrict__ acc) {
    __shared__ float sa[512], sb[512], sc[512];
    __shared__ float red[64];
    const int r = blockIdx.x;      // 0..63
    const int s = threadIdx.x;     // 0..63
    for (int i = s; i < 512; i += 64) {
        sa[i] = A[i * R_DIM + r];
        sb[i] = B[i * R_DIM + r];
        sc[i] = C[i * R_DIM + r];
    }
    __syncthreads();
    float ga = 0.f, gb = 0.f, gc = 0.f;
    for (int i = 0; i < 512; ++i) {
        ga = fmaf(sa[i], A[i * R_DIM + s], ga);
        gb = fmaf(sb[i], B[i * R_DIM + s], gb);
        gc = fmaf(sc[i], C[i * R_DIM + s], gc);
    }
    red[s] = ga * gb * gc;
    __syncthreads();
    for (int off = 32; off; off >>= 1) {
        if (s < off) red[s] += red[s + off];
        __syncthreads();
    }
    if (s == 0) atomicAdd(acc + 2, red[0]);
}

// ---- fused main kernel: sum X^2  and  sum(X .* recon) ---------------------
__global__ __launch_bounds__(256)
void cp_fused_kernel(const float* __restrict__ Xd,
                     const float* __restrict__ Ad,   // data   [512][64]
                     const float* __restrict__ Bd,   // nmode1 [512][64]
                     const __bf16* __restrict__ CtHi,
                     const __bf16* __restrict__ CtLo,
                     float* __restrict__ acc) {
    const int lane   = threadIdx.x & 31;
    const int wave   = threadIdx.x >> 5;
    const int rowBase = blockIdx.x * ROWS_PER_BLK + wave * ROWS_PER_WAVE;
    const int m16    = lane & 15;
    const bool hiHalf = (lane >= 16);

    // A-fragment (contiguous k order, matching B): lane m16 owns row rowBase+m16,
    // loads 16 contiguous K values {koff..koff+15}, koff = 0 / 16.
    const float* xrow = Xd + (size_t)(rowBase + m16) * K_DIM;
    const int koff = hiHalf ? 16 : 0;
    // B-fragment (contiguous k order, matching A): lane owns column n = nt*16+m16,
    // 16 contiguous K values starting at k0 (+16 for the high lane half).
    const int bko = hiHalf ? 16 : 0;

    v8f accv[4];
#pragma unroll
    for (int nt = 0; nt < 4; ++nt) accv[nt] = (v8f)0.0f;
    float x2 = 0.f;

    for (int kc = 0; kc < K_DIM / 32; ++kc) {
        const int k0 = kc * 32;
        const float4* p = (const float4*)(xrow + k0 + koff);
        const float4* q = (const float4*)(xrow + k0 + koff + 8);
        float4 x0 = p[0];
        float4 x1 = p[1];
        float4 y0 = q[0];
        float4 y1 = q[1];
        float xs[16] = { x0.x, x0.y, x0.z, x0.w,  x1.x, x1.y, x1.z, x1.w,
                         y0.x, y0.y, y0.z, y0.w,  y1.x, y1.y, y1.z, y1.w };
        v16bf ah, al;
#pragma unroll
        for (int t = 0; t < 16; ++t) {
            float v = xs[t];
            x2 = fmaf(v, v, x2);            // each X element touched exactly once
            __bf16 h = (__bf16)v;
            ah[t] = h;
            al[t] = (__bf16)(v - (float)h); // residual
        }
#pragma unroll
        for (int nt = 0; nt < 4; ++nt) {
            const size_t boff = (size_t)(nt * 16 + m16) * K_DIM + k0 + bko;
            v16bf bh = *(const v16bf*)(CtHi + boff);
            v16bf bl = *(const v16bf*)(CtLo + boff);
            v8f c = accv[nt];
            c = __builtin_amdgcn_wmma_f32_16x16x32_bf16(false, ah, false, bh,
                                                        (short)0, c, false, false);
            c = __builtin_amdgcn_wmma_f32_16x16x32_bf16(false, ah, false, bl,
                                                        (short)0, c, false, false);
            c = __builtin_amdgcn_wmma_f32_16x16x32_bf16(false, al, false, bh,
                                                        (short)0, c, false, false);
            accv[nt] = c;
        }
    }

    // Epilogue: D[m, r] holds (X_row_m . C[:,r]); weight by A[i,r]*B[j,r].
    // All 16 rows of this wave share one i (rows are 16-aligned, J=512).
    const int i_idx = rowBase >> 9;
    const int j0    = rowBase & 511;
    const int mo    = hiHalf ? 8 : 0;     // C/D layout: VGPR v -> M = v (+8)
    float xr = 0.f;
#pragma unroll
    for (int nt = 0; nt < 4; ++nt) {
        const int r  = nt * 16 + m16;     // C/D layout: N = lane % 16
        const float dw = Ad[i_idx * R_DIM + r];
#pragma unroll
        for (int v = 0; v < 8; ++v) {
            const float w = dw * Bd[(j0 + mo + v) * R_DIM + r];
            xr = fmaf(accv[nt][v], w, xr);
        }
    }

    // wave32 reduction, then one atomic per wave
    for (int off = 16; off; off >>= 1) {
        xr += __shfl_down(xr, off, 32);
        x2 += __shfl_down(x2, off, 32);
    }
    if (lane == 0) {
        atomicAdd(acc + 1, xr);
        atomicAdd(acc + 0, x2);
    }
}

// ---- combine --------------------------------------------------------------
__global__ void combine_kernel(const float* __restrict__ acc,
                               float* __restrict__ out) {
    out[0] = acc[0] - 2.0f * acc[1] + acc[2];
}

extern "C" void kernel_launch(void* const* d_in, const int* in_sizes, int n_in,
                              void* d_out, int out_size, void* d_ws, size_t ws_size,
                              hipStream_t stream) {
    const float* A = (const float*)d_in[0];   // data   (512, 64)
    const float* B = (const float*)d_in[1];   // nmode1 (512, 64)
    const float* C = (const float*)d_in[2];   // nmode2 (512, 64)
    const float* X = (const float*)d_in[3];   // X      (512, 512, 512)

    float*  acc  = (float*)d_ws;
    __bf16* CtHi = (__bf16*)((char*)d_ws + WS_CT_OFF);
    __bf16* CtLo = (__bf16*)((char*)d_ws + WS_CT_OFF + CT_BYTES);

    hipMemsetAsync(d_ws, 0, WS_CT_OFF, stream);   // zero the accumulators

    ct_split_kernel<<<(R_DIM * K_DIM) / 256, 256, 0, stream>>>(C, CtHi, CtLo);
    gram_kernel<<<R_DIM, 64, 0, stream>>>(A, B, C, acc);
    cp_fused_kernel<<<IJ_ROWS / ROWS_PER_BLK, 256, 0, stream>>>(X, A, B, CtHi, CtLo, acc);
    combine_kernel<<<1, 1, 0, stream>>>(acc, (float*)d_out);
}
